// GraphHyperbolicVisitEncoderGlobal_86217173500282
// MI455X (gfx1250) — matrix operand order repaired and checked by reading, not compiled, not run on hardware
//
#include <hip/hip_runtime.h>
#include <hip/hip_bf16.h>
#include <cstdint>

// ---------------------------------------------------------------------------
// Types for CDNA5 WMMA (wave32): 16x16x32 bf16 -> f32
// ---------------------------------------------------------------------------
typedef __attribute__((ext_vector_type(16))) __bf16 v16bf;
typedef __attribute__((ext_vector_type(8)))  float  v8f;
typedef __attribute__((ext_vector_type(4)))  unsigned int v4u;
typedef __attribute__((ext_vector_type(8)))  int v8i_;
typedef __attribute__((ext_vector_type(4)))  int v4i_;

#define V_CODES 4096
#define D_DIM   128
#define K_SC    3
#define H_HEADS 4
#define HD      32
#define FF_DIM  128
#define NVIS    8192
#define MAXC    32

// Tensor Data Mover availability. On the gfx1250 device pass we REQUIRE the
// TDM builtins so that a successful compile certifies the async-tensor path
// is actually in the emitted code (no silent fallback).
#if defined(__has_builtin)
#if __has_builtin(__builtin_amdgcn_tensor_load_to_lds) && \
    __has_builtin(__builtin_amdgcn_s_wait_tensorcnt)
#define USE_TDM 1
#endif
#endif
#ifndef USE_TDM
#define USE_TDM 0
#endif
#if defined(__gfx1250__) && !USE_TDM
#error "gfx1250 device pass without tensor_load_to_lds/s_wait_tensorcnt builtins"
#endif

// ---------------------------------------------------------------------------
// WMMA wrapper
// ---------------------------------------------------------------------------
__device__ __forceinline__ v8f wmma_bf16(v16bf a, v16bf b, v8f c) {
  return __builtin_amdgcn_wmma_f32_16x16x32_bf16(
      /*neg_a=*/false, a, /*neg_b=*/false, b,
      /*c_mod=*/(short)0, c, /*reuse_a=*/false, /*reuse_b=*/false);
}

// ---------------------------------------------------------------------------
// Fragment loaders (wave32; A/B in this file's contiguous k order, not the ISA table order)
// A (16x32): lane L<16 row M=L: elem e -> K e
//            lane L>=16 row M=L-16: elem e -> K 16+e
// B (32x16): lane L<16 col N=L, elem e -> K e; lanes>=16 -> K 16+e
// C (16x16 f32): lane L<16 col N=L, vgpr r -> M=r; lanes>=16 -> M=8+r
// ---------------------------------------------------------------------------
__device__ __forceinline__ v16bf load_a_frag_g(const float* __restrict__ A,
                                               int lda, int m0, int k0) {
  const int lane = threadIdx.x & 31;
  const int hf   = lane >> 4;
  const float* p = A + (size_t)(m0 + (lane & 15)) * lda + k0 + hf * 16;
  v16bf a;
#pragma unroll
  for (int e = 0; e < 16; ++e) a[e] = (__bf16)p[e];
  return a;
}

// A fragment from an fp32 LDS tile with row stride `ld`, k-offset k0
__device__ __forceinline__ v16bf load_a_frag_lds_f32(const float* __restrict__ As,
                                                     int ld, int k0) {
  const int lane = threadIdx.x & 31;
  const int hf   = lane >> 4;
  const float* p = As + (lane & 15) * ld + k0 + hf * 16;
  v16bf a;
#pragma unroll
  for (int e = 0; e < 16; ++e) a[e] = (__bf16)p[e];
  return a;
}

// A fragment from a bf16 LDS tile (16x32 row-major) — used by flash attention
__device__ __forceinline__ v16bf load_a_frag_lds(const __bf16* __restrict__ As) {
  const int lane = threadIdx.x & 31;
  const int hf   = lane >> 4;
  const __bf16* p = As + (lane & 15) * 32 + hf * 16;
  v16bf a;
#pragma unroll
  for (int e = 0; e < 16; ++e) a[e] = p[e];
  return a;
}

// B[k][n] row-major with leading dim ldb (strided per-lane column read)
__device__ __forceinline__ v16bf load_b_frag_g(const float* __restrict__ B,
                                               int ldb, int k0, int n0) {
  const int lane = threadIdx.x & 31;
  const int hf   = lane >> 4;
  const float* p = B + (size_t)(k0 + hf * 16) * ldb + n0 + (lane & 15);
  v16bf b;
#pragma unroll
  for (int e = 0; e < 16; ++e) b[e] = (__bf16)p[(size_t)e * ldb];
  return b;
}

// B[k][n] = W[n][k] where W is [N x K] row-major (contiguous per-lane read)
__device__ __forceinline__ v16bf load_bT_frag_g(const float* __restrict__ W,
                                                int ldw, int k0, int n0) {
  const int lane = threadIdx.x & 31;
  const int hf   = lane >> 4;
  const float* p = W + (size_t)(n0 + (lane & 15)) * ldw + k0 + hf * 16;
  v16bf b;
#pragma unroll
  for (int e = 0; e < 16; ++e) b[e] = (__bf16)p[e];
  return b;
}

// ---------------------------------------------------------------------------
// Cross-lane reductions via ds_swizzle (group-of-32 mode: and=0x1f, xor=m)
// ---------------------------------------------------------------------------
__device__ __forceinline__ float red_max16(float v) {
  v = fmaxf(v, __int_as_float(__builtin_amdgcn_ds_swizzle(__float_as_int(v), (1 << 10) | 0x1f)));
  v = fmaxf(v, __int_as_float(__builtin_amdgcn_ds_swizzle(__float_as_int(v), (2 << 10) | 0x1f)));
  v = fmaxf(v, __int_as_float(__builtin_amdgcn_ds_swizzle(__float_as_int(v), (4 << 10) | 0x1f)));
  v = fmaxf(v, __int_as_float(__builtin_amdgcn_ds_swizzle(__float_as_int(v), (8 << 10) | 0x1f)));
  return v;
}
__device__ __forceinline__ float red_sum16(float v) {
  v += __int_as_float(__builtin_amdgcn_ds_swizzle(__float_as_int(v), (1 << 10) | 0x1f));
  v += __int_as_float(__builtin_amdgcn_ds_swizzle(__float_as_int(v), (2 << 10) | 0x1f));
  v += __int_as_float(__builtin_amdgcn_ds_swizzle(__float_as_int(v), (4 << 10) | 0x1f));
  v += __int_as_float(__builtin_amdgcn_ds_swizzle(__float_as_int(v), (8 << 10) | 0x1f));
  return v;
}
__device__ __forceinline__ float red_sum32(float v) {
  v = red_sum16(v);
  v += __int_as_float(__builtin_amdgcn_ds_swizzle(__float_as_int(v), (16 << 10) | 0x1f));
  return v;
}

__device__ __forceinline__ float gelu_tanh(float x) {
  float x3 = x * x * x;
  return 0.5f * x * (1.0f + tanhf(0.7978845608028654f * (x + 0.044715f * x3)));
}

// ---------------------------------------------------------------------------
// Tensor Data Mover: 2D tile (rows x cols fp32) global -> LDS.
// D# layout per CDNA5 ISA (group0: flags/lds/global/type,
// group1: data_size + dims/strides). Groups 2/3 zero (<=2D tensor).
// Issued by wave 0 only; completion via s_wait_tensorcnt.
// ---------------------------------------------------------------------------
#if USE_TDM
__device__ __forceinline__ unsigned lds_byte_off(const void* p) {
  // generic LDS address: addr[31:0] is the LDS offset (aperture in high bits)
  return (unsigned)(unsigned long long)(uintptr_t)p;
}

__device__ __forceinline__ void tdm_load_tile_f32(const float* gptr,
                                                  unsigned lds_off,
                                                  int rows, int cols,
                                                  int ld_elems) {
  unsigned long long ga = (unsigned long long)(uintptr_t)gptr;
  v4u g0;
  g0[0] = 1u;                                        // count=1 (valid), user mode
  g0[1] = lds_off;                                   // lds_addr (bytes)
  g0[2] = (unsigned)(ga & 0xffffffffu);              // global_addr[31:0]
  g0[3] = (unsigned)((ga >> 32) & 0x1ffffffu)        // global_addr[56:32]
          | (2u << 30);                              // type = 2 ("image")
  const unsigned td0 = (unsigned)ld_elems;           // tensor_dim0 (elems)
  const unsigned td1 = (unsigned)V_CODES;            // tensor_dim1
  const unsigned long long s0 = (unsigned long long)ld_elems;  // dim0 stride
  v8i_ g1;
  g1[0] = (int)(2u << 16);                           // data_size=2 -> 4 bytes
  g1[1] = (int)((td0 & 0xffffu) << 16);              // tensor_dim0[15:0]
  g1[2] = (int)((td0 >> 16) | ((td1 & 0xffffu) << 16));
  g1[3] = (int)((td1 >> 16) | ((unsigned)cols << 16));   // tile_dim0 = cols
  g1[4] = (int)((unsigned)rows);                     // tile_dim1 = rows, tile_dim2=0
  g1[5] = (int)(s0 & 0xffffffffu);                   // tensor_dim0_stride[31:0]
  g1[6] = (int)((s0 >> 32) & 0xffffu);               // stride[47:32], stride1=0
  g1[7] = 0;
  const v4i_ zz = {0, 0, 0, 0};
#if __clang_major__ >= 23
  const v8i_ zz8 = {0, 0, 0, 0, 0, 0, 0, 0};
  __builtin_amdgcn_tensor_load_to_lds(g0, g1, zz, zz, zz8, 0);
#else
  __builtin_amdgcn_tensor_load_to_lds(g0, g1, zz, zz, 0);
#endif
}
#endif  // USE_TDM

// ---------------------------------------------------------------------------
// Generic GEMM: C[M x N] = A[M x K] @ B (+bias, +gelu, +residual)
// Block = 256 threads (8 waves), one 16-row strip per block. The A strip is
// staged into LDS in 16x64 fp32 chunks, double-buffered via the Tensor Data
// Mover (TENSORcnt), so the 201 MB `kernels` stream overlaps the WMMAs and is
// read from HBM exactly once. Two v_wmma_f32_16x16x32_bf16 per chunk.
// Requires N % 128 == 0, K % 64 == 0, M = 16 * gridDim.x.
// EPI: 0=none 1=+bias 2=+bias,gelu 3=+bias,+residual
// ---------------------------------------------------------------------------
template <bool TRANSB, int EPI>
__global__ __launch_bounds__(256) void gemm16_kernel(
    const float* __restrict__ A, int lda,
    const float* __restrict__ B, int ldb,
    const float* __restrict__ bias,
    const float* __restrict__ res, int ldres,
    float* __restrict__ C, int ldc, int N, int K) {
  __shared__ float Af[2][16 * 64];  // 8 KB double buffer
  const int tid  = threadIdx.x;
  const int wv   = tid >> 5;
  const int lane = tid & 31;
  const int hf   = lane >> 4;
  const int col  = lane & 15;
  const int m0   = blockIdx.x * 16;
  const int nk   = K >> 6;  // 64-wide k-chunks

  for (int n0 = wv * 16; n0 < N; n0 += 128) {
    v8f acc = {0.f, 0.f, 0.f, 0.f, 0.f, 0.f, 0.f, 0.f};
#if USE_TDM
    if (wv == 0)  // prologue: DMA chunk 0 into buffer 0
      tdm_load_tile_f32(A + (size_t)m0 * lda, lds_byte_off(&Af[0][0]), 16, 64, lda);
#endif
    for (int i = 0; i < nk; ++i) {
      const int k0 = i << 6;
      const float* Ab = &Af[i & 1][0];
#if USE_TDM
      if (wv == 0) {
        if (i + 1 < nk) {
          // overlap: DMA next chunk while this one is consumed
          tdm_load_tile_f32(A + (size_t)m0 * lda + (k0 + 64),
                            lds_byte_off(&Af[(i + 1) & 1][0]), 16, 64, lda);
          __builtin_amdgcn_s_wait_tensorcnt(1);  // chunk i landed (in-order)
        } else {
          __builtin_amdgcn_s_wait_tensorcnt(0);
        }
      }
      __syncthreads();
#else
      {  // host-pass / non-TDM fallback: cooperative synchronous staging
        float* dst = &Af[i & 1][0];
#pragma unroll
        for (int e = tid; e < 16 * 64; e += 256)
          dst[e] = A[(size_t)(m0 + (e >> 6)) * lda + k0 + (e & 63)];
        if (k0 + 64 < K)
          __builtin_prefetch(&A[(size_t)(m0 + (tid >> 5)) * lda + k0 + 64 + (tid & 31)], 0, 1);
      }
      __syncthreads();
#endif
      v16bf a0 = load_a_frag_lds_f32(Ab, 64, 0);
      v16bf b0 = TRANSB ? load_bT_frag_g(B, ldb, k0, n0)
                        : load_b_frag_g(B, ldb, k0, n0);
      acc = wmma_bf16(a0, b0, acc);
      v16bf a1 = load_a_frag_lds_f32(Ab, 64, 32);
      v16bf b1 = TRANSB ? load_bT_frag_g(B, ldb, k0 + 32, n0)
                        : load_b_frag_g(B, ldb, k0 + 32, n0);
      acc = wmma_bf16(a1, b1, acc);
      __syncthreads();
    }
#pragma unroll
    for (int r = 0; r < 8; ++r) {
      const int row = m0 + r + hf * 8;
      const int cc  = n0 + col;
      float v = acc[r];
      if (EPI >= 1) v += bias[cc];
      if (EPI == 2) v = gelu_tanh(v);
      if (EPI == 3) v += res[(size_t)row * ldres + cc];
      C[(size_t)row * ldc + cc] = v;
    }
  }
}

// ---------------------------------------------------------------------------
// Flash attention over V=4096 tokens. Block = 128 threads (4 waves), one wave
// per head, one 16-query block per CTA. hd=32 => Q@K^T is one WMMA K-step.
// Online softmax with ds_swizzle row reductions; P-tile re-layout
// (C-frag -> A-frag) through 1 KB wave-private LDS. No score materialization.
// ---------------------------------------------------------------------------
__global__ __launch_bounds__(128) void flash_attn_kernel(
    const float* __restrict__ qkv, float* __restrict__ out) {
  __shared__ __bf16 Plds[4][16 * 32];
  const int wv   = threadIdx.x >> 5;  // head index
  const int lane = threadIdx.x & 31;
  const int hf   = lane >> 4;
  const int col  = lane & 15;
  const int h    = wv;
  const int q0   = blockIdx.x * 16;
  const float scale = 0.17677669529663687f;  // 1/sqrt(32)

  v16bf aq = load_a_frag_g(qkv + h * HD, 3 * D_DIM, q0, 0);

  float mrow[8], lrow[8];
  v8f o0 = {0.f, 0.f, 0.f, 0.f, 0.f, 0.f, 0.f, 0.f};
  v8f o1 = {0.f, 0.f, 0.f, 0.f, 0.f, 0.f, 0.f, 0.f};
#pragma unroll
  for (int r = 0; r < 8; ++r) { mrow[r] = -1e30f; lrow[r] = 0.f; }

  __bf16* P = &Plds[wv][0];
  const float* Kbase = qkv + D_DIM + h * HD;
  const v8f zero = {0.f, 0.f, 0.f, 0.f, 0.f, 0.f, 0.f, 0.f};

  for (int t0 = 0; t0 < V_CODES; t0 += 32) {
    v16bf bk0 = load_bT_frag_g(Kbase, 3 * D_DIM, 0, t0);
    v16bf bk1 = load_bT_frag_g(Kbase, 3 * D_DIM, 0, t0 + 16);
    v8f s0 = wmma_bf16(aq, bk0, zero);
    v8f s1 = wmma_bf16(aq, bk1, zero);

#pragma unroll
    for (int r = 0; r < 8; ++r) {
      float a0 = s0[r] * scale;
      float a1 = s1[r] * scale;
      float mx = red_max16(fmaxf(a0, a1));
      float mnew = fmaxf(mrow[r], mx);
      float corr = __expf(mrow[r] - mnew);
      float p0 = __expf(a0 - mnew);
      float p1 = __expf(a1 - mnew);
      float rs = red_sum16(p0 + p1);
      lrow[r] = lrow[r] * corr + rs;
      o0[r] *= corr;
      o1[r] *= corr;
      mrow[r] = mnew;
      P[(r + hf * 8) * 32 + col]      = (__bf16)p0;
      P[(r + hf * 8) * 32 + 16 + col] = (__bf16)p1;
    }
    __syncthreads();
    v16bf ap = load_a_frag_lds(P);
    const float* vbase = qkv + (size_t)t0 * (3 * D_DIM) + 2 * D_DIM + h * HD;
    v16bf bv0 = load_b_frag_g(vbase, 3 * D_DIM, 0, 0);
    v16bf bv1 = load_b_frag_g(vbase, 3 * D_DIM, 0, 16);
    o0 = wmma_bf16(ap, bv0, o0);
    o1 = wmma_bf16(ap, bv1, o1);
    __syncthreads();
  }

#pragma unroll
  for (int r = 0; r < 8; ++r) {
    float inv = 1.0f / lrow[r];
    int row = q0 + r + hf * 8;
    out[(size_t)row * D_DIM + h * HD + col]      = o0[r] * inv;
    out[(size_t)row * D_DIM + h * HD + 16 + col] = o1[r] * inv;
  }
}

// ---------------------------------------------------------------------------
// logmap0: Z = artanh(clip(||x||)) * x / max(||x||, eps). One row per block.
// ---------------------------------------------------------------------------
__global__ __launch_bounds__(128) void logmap0_kernel(
    const float* __restrict__ X, float* __restrict__ Z) {
  __shared__ float red[4];
  const int row = blockIdx.x, d = threadIdx.x;
  float v = X[(size_t)row * D_DIM + d];
  float s = red_sum32(v * v);
  if ((threadIdx.x & 31) == 0) red[threadIdx.x >> 5] = s;
  __syncthreads();
  float norm = sqrtf(red[0] + red[1] + red[2] + red[3]);
  float nc = fminf(fmaxf(norm, 1e-7f), 1.0f - 1e-5f);
  float sc = atanhf(nc) / fmaxf(norm, 1e-7f);
  Z[(size_t)row * D_DIM + d] = v * sc;
}

// ---------------------------------------------------------------------------
// LayerNorm over D=128, one row per block.
// ---------------------------------------------------------------------------
__global__ __launch_bounds__(128) void layernorm_kernel(
    const float* __restrict__ x, const float* __restrict__ g,
    const float* __restrict__ b, float* __restrict__ y) {
  __shared__ float red[4];
  const int row = blockIdx.x, d = threadIdx.x;
  float v = x[(size_t)row * D_DIM + d];
  float s = red_sum32(v);
  if ((threadIdx.x & 31) == 0) red[threadIdx.x >> 5] = s;
  __syncthreads();
  float mean = (red[0] + red[1] + red[2] + red[3]) * (1.0f / D_DIM);
  float dv = v - mean;
  __syncthreads();
  float s2 = red_sum32(dv * dv);
  if ((threadIdx.x & 31) == 0) red[threadIdx.x >> 5] = s2;
  __syncthreads();
  float var = (red[0] + red[1] + red[2] + red[3]) * (1.0f / D_DIM);
  y[(size_t)row * D_DIM + d] = dv * rsqrtf(var + 1e-5f) * g[d] + b[d];
}

// ---------------------------------------------------------------------------
// Per-visit masked mean pool (PAD = code 0). One visit per block, thread = dim.
// ---------------------------------------------------------------------------
__global__ __launch_bounds__(128) void pool_kernel(
    const int* __restrict__ visits, const float* __restrict__ Hf,
    float* __restrict__ out) {
  const int vi = blockIdx.x, d = threadIdx.x;
  const int* row = visits + (size_t)vi * MAXC;
  float sum = 0.f;
  int cnt = 0;
#pragma unroll 4
  for (int j = 0; j < MAXC; ++j) {
    int c = row[j];
    if (c != 0) { sum += Hf[(size_t)c * D_DIM + d]; ++cnt; }
  }
  out[(size_t)vi * D_DIM + d] = (cnt > 0) ? sum / (float)cnt : 0.f;
}

// ---------------------------------------------------------------------------
// Host orchestration (graph-capture safe: stream launches only)
// ---------------------------------------------------------------------------
extern "C" void kernel_launch(void* const* d_in, const int* in_sizes, int n_in,
                              void* d_out, int out_size, void* d_ws, size_t ws_size,
                              hipStream_t stream) {
  (void)in_sizes; (void)n_in; (void)out_size; (void)ws_size;
  const int*   visits  = (const int*)  d_in[0];
  const float* X_hyp   = (const float*)d_in[1];
  const float* kernels = (const float*)d_in[2];
  const float* proj_W  = (const float*)d_in[3];
  const float* proj_b  = (const float*)d_in[4];
  const float* Wqkv    = (const float*)d_in[5];
  const float* bqkv    = (const float*)d_in[6];
  const float* Wo      = (const float*)d_in[7];
  const float* bo      = (const float*)d_in[8];
  const float* W1      = (const float*)d_in[9];
  const float* b1      = (const float*)d_in[10];
  const float* W2      = (const float*)d_in[11];
  const float* b2      = (const float*)d_in[12];
  const float* ln1_g   = (const float*)d_in[13];
  const float* ln1_b   = (const float*)d_in[14];
  const float* ln2_g   = (const float*)d_in[15];
  const float* ln2_b   = (const float*)d_in[16];

  float* ws   = (float*)d_ws;
  float* Z0   = ws;                                  // [4096 x 128]
  float* Zcat = Z0   + (size_t)V_CODES * D_DIM;      // [4096 x 384]
  float* Hbuf = Zcat + (size_t)V_CODES * 3 * D_DIM;  // [4096 x 128]
  float* qkvb = Hbuf + (size_t)V_CODES * D_DIM;      // [4096 x 384]
  float* attn = qkvb + (size_t)V_CODES * 3 * D_DIM;  // [4096 x 128]
  float* ffb  = attn + (size_t)V_CODES * D_DIM;      // [4096 x 128]
  float* tmpb = ffb  + (size_t)V_CODES * FF_DIM;     // [4096 x 128]

  const int MB = V_CODES / 16;  // 256 blocks (16-row strips)

  // 1) hyperbolic log map
  logmap0_kernel<<<V_CODES, 128, 0, stream>>>(X_hyp, Z0);

  // 2) diffusion: Zcat[:, s*128:(s+1)*128] = kernels[s] @ Z0  (HBM-bound)
  for (int s = 0; s < K_SC; ++s) {
    gemm16_kernel<false, 0><<<MB, 256, 0, stream>>>(
        kernels + (size_t)s * V_CODES * V_CODES, V_CODES,
        Z0, D_DIM, nullptr, nullptr, 0,
        Zcat + s * D_DIM, 3 * D_DIM, D_DIM, V_CODES);
  }

  // 3) projection: H = Zcat @ proj_W.T + proj_b
  gemm16_kernel<true, 1><<<MB, 256, 0, stream>>>(
      Zcat, 3 * D_DIM, proj_W, 3 * D_DIM, proj_b, nullptr, 0,
      Hbuf, D_DIM, D_DIM, 3 * D_DIM);

  // 4) attention stack
  for (int l = 0; l < 2; ++l) {
    gemm16_kernel<true, 1><<<MB, 256, 0, stream>>>(
        Hbuf, D_DIM, Wqkv + (size_t)l * 3 * D_DIM * D_DIM, D_DIM,
        bqkv + (size_t)l * 3 * D_DIM, nullptr, 0,
        qkvb, 3 * D_DIM, 3 * D_DIM, D_DIM);

    flash_attn_kernel<<<MB, 128, 0, stream>>>(qkvb, attn);

    gemm16_kernel<true, 3><<<MB, 256, 0, stream>>>(
        attn, D_DIM, Wo + (size_t)l * D_DIM * D_DIM, D_DIM,
        bo + (size_t)l * D_DIM, Hbuf, D_DIM,
        tmpb, D_DIM, D_DIM, D_DIM);
    layernorm_kernel<<<V_CODES, 128, 0, stream>>>(
        tmpb, ln1_g + (size_t)l * D_DIM, ln1_b + (size_t)l * D_DIM, Hbuf);

    gemm16_kernel<true, 2><<<MB, 256, 0, stream>>>(
        Hbuf, D_DIM, W1 + (size_t)l * FF_DIM * D_DIM, D_DIM,
        b1 + (size_t)l * FF_DIM, nullptr, 0,
        ffb, FF_DIM, FF_DIM, D_DIM);
    gemm16_kernel<true, 3><<<MB, 256, 0, stream>>>(
        ffb, FF_DIM, W2 + (size_t)l * D_DIM * FF_DIM, FF_DIM,
        b2 + (size_t)l * D_DIM, Hbuf, D_DIM,
        tmpb, D_DIM, D_DIM, FF_DIM);
    layernorm_kernel<<<V_CODES, 128, 0, stream>>>(
        tmpb, ln2_g + (size_t)l * D_DIM, ln2_b + (size_t)l * D_DIM, Hbuf);
  }

  // 5) per-visit masked mean pooling
  pool_kernel<<<NVIS, 128, 0, stream>>>(visits, Hbuf, (float*)d_out);
}
